// TMIRIM_46110768890274
// MI455X (gfx1250) — matrix operand order repaired and checked by reading, not compiled, not run on hardware
//
#include <hip/hip_runtime.h>
#include <hip/hip_bf16.h>
#include <math.h>

typedef __attribute__((ext_vector_type(16))) _Float16 v16h;
typedef __attribute__((ext_vector_type(8)))  _Float16 v8h;
typedef __attribute__((ext_vector_type(8)))  float    v8f;

#define NN   4096
#define MM   3
#define FF   64
#define HH   64
#define BETA 1.0f

// ---------------------------------------------------------------------------
// zero the global S accumulator (graph-capture-safe, no memset)
// ---------------------------------------------------------------------------
__global__ void zero_accum(float* p) { p[0] = 0.0f; }

// ---------------------------------------------------------------------------
// prep: h = h_tilde @ W_proj + b ; w = T@alpha ; g = w-mix of h
// store h,g as f16 in k-major [k][n][64] layout for the WMMA GEMM.
// one block per n, 64 threads (one per feature d).
// ---------------------------------------------------------------------------
__global__ __launch_bounds__(64) void prep_kernel(
    const float* __restrict__ h_tilde,  // [N][M][F]
    const float* __restrict__ T,        // [M][M][H]
    const float* __restrict__ alpha,    // [H]
    const float* __restrict__ Wp,       // [F][H]
    const float* __restrict__ bp,       // [H]
    _Float16* __restrict__ hK,          // [M][N][H]
    _Float16* __restrict__ gK)          // [M][N][H]
{
    __shared__ float sh_h[MM][FF];
    __shared__ float sh_alpha[HH];
    __shared__ float sh_w[MM * MM];

    const int n = blockIdx.x;
    const int d = threadIdx.x;

    sh_alpha[d] = alpha[d];
    #pragma unroll
    for (int m = 0; m < MM; ++m)
        sh_h[m][d] = h_tilde[((size_t)n * MM + m) * FF + d];
    __syncthreads();

    if (d < MM * MM) {
        const int k = d / MM, m = d % MM;
        float acc = 0.f;
        #pragma unroll 4
        for (int l = 0; l < HH; ++l)
            acc += T[((size_t)k * MM + m) * HH + l] * sh_alpha[l];
        sh_w[d] = acc;
    }
    __syncthreads();

    float hv[MM];
    #pragma unroll
    for (int m = 0; m < MM; ++m) {
        float acc = bp[d];
        #pragma unroll 4
        for (int f = 0; f < FF; ++f)
            acc += sh_h[m][f] * Wp[(size_t)f * HH + d];   // coalesced over d
        hv[m] = acc;
        hK[((size_t)m * NN + n) * HH + d] = (_Float16)acc;
    }
    #pragma unroll
    for (int k = 0; k < MM; ++k) {
        float acc = sh_w[k * MM + 0] * hv[0]
                  + sh_w[k * MM + 1] * hv[1]
                  + sh_w[k * MM + 2] * hv[2];
        gK[((size_t)k * NN + n) * HH + d] = (_Float16)acc;
    }
}

__device__ __forceinline__ float lrelu(float x) { return x > 0.f ? x : 0.01f * x; }

// ---------------------------------------------------------------------------
// fused GEMM: r[i,j,k] = h_k[i,:] . g_k[j,:]  via v_wmma_f32_16x16x32_f16
// pass 0: accumulate Sum(S) (global) via wave-reduce + atomicAdd
// pass 1: recompute, mask vs tau, softmax over k, write R [N][N][3]
// 256 threads = 8 waves/block, one 16x16 (x3 k) tile per wave.
// ---------------------------------------------------------------------------
__global__ __launch_bounds__(256) void fused_gemm(
    const _Float16* __restrict__ hK,    // [M][N][H]
    const _Float16* __restrict__ gK,    // [M][N][H]
    float* __restrict__ sumS,           // accumulator / tau source
    float* __restrict__ out,            // [N][N][M]
    int pass)
{
    const int wave = threadIdx.x >> 5;
    const int lane = threadIdx.x & 31;
    const int tile = blockIdx.x * 8 + wave;     // 65536 tiles total
    const int ti = tile >> 8;                   // 256 tiles per dim
    const int tj = tile & 255;
    const int i0 = ti * 16;
    const int j0 = tj * 16;
    const int sel  = lane >> 4;                 // half-wave select
    const int lrow = lane & 15;

    v8f c[MM];
    #pragma unroll
    for (int k = 0; k < MM; ++k) {
        v8f acc = {};
        const _Float16* aBase = hK + ((size_t)k * NN + (i0 + lrow)) * HH;
        const _Float16* bBase = gK + ((size_t)k * NN + (j0 + lrow)) * HH;
        #pragma unroll
        for (int kk = 0; kk < 2; ++kk) {
            // A 16x32 f16: lanes 0-15 -> K[0..15]; lanes 16-31 -> K[16..31]
            v16h a = *(const v16h*)(aBase + kk * 32 + sel * 16);
            // B 32x16 (= g^T): lane n holds column n, lanes 0-15 K[0..15], 16-31 K[16..31]
            v16h b = *(const v16h*)(bBase + kk * 32 + sel * 16);
            acc = __builtin_amdgcn_wmma_f32_16x16x32_f16(
                      false, a, false, b, (short)0, acc, false, false);
        }
        c[k] = acc;
    }

    if (pass == 0) {
        float local = 0.f;
        #pragma unroll
        for (int e = 0; e < 8; ++e)
            local += lrelu(c[0][e]) + lrelu(c[1][e]) + lrelu(c[2][e]);
        // wave32 reduction
        #pragma unroll
        for (int off = 16; off > 0; off >>= 1)
            local += __shfl_down(local, off, 32);
        if (lane == 0) atomicAdd(sumS, local);
    } else {
        const float tau = BETA * sumS[0] * (1.0f / ((float)NN * (float)NN));
        #pragma unroll
        for (int e = 0; e < 8; ++e) {
            const float r0 = lrelu(c[0][e]);
            const float r1 = lrelu(c[1][e]);
            const float r2 = lrelu(c[2][e]);
            const float S  = r0 + r1 + r2;
            const int i = i0 + e + sel * 8;   // C layout: M = e (+8 for upper half-wave)
            const int j = j0 + lrow;          // N = lane % 16
            float* o = out + ((size_t)i * NN + j) * MM;
            if (S > tau) {
                const float mx = fmaxf(r0, fmaxf(r1, r2));
                const float e0 = __expf(r0 - mx);
                const float e1 = __expf(r1 - mx);
                const float e2 = __expf(r2 - mx);
                const float inv = 1.f / (e0 + e1 + e2);
                o[0] = e0 * inv; o[1] = e1 * inv; o[2] = e2 * inv;
            } else {
                o[0] = 0.f; o[1] = 0.f; o[2] = 0.f;
            }
        }
    }
}

// ---------------------------------------------------------------------------
extern "C" void kernel_launch(void* const* d_in, const int* in_sizes, int n_in,
                              void* d_out, int out_size, void* d_ws, size_t ws_size,
                              hipStream_t stream) {
    const float* h_tilde = (const float*)d_in[0];   // [4096][3][64]
    const float* T       = (const float*)d_in[1];   // [3][3][64]
    const float* alpha   = (const float*)d_in[2];   // [64]
    const float* Wp      = (const float*)d_in[3];   // [64][64]
    const float* bp      = (const float*)d_in[4];   // [64]
    float* out = (float*)d_out;                     // [4096][4096][3]

    // workspace layout: [0..255] sum accumulator | hK (f16) | gK (f16)
    char* ws = (char*)d_ws;
    float*     sumS = (float*)ws;
    _Float16*  hK   = (_Float16*)(ws + 256);
    const size_t hgBytes = (size_t)MM * NN * HH * sizeof(_Float16); // 1.5 MB
    _Float16*  gK   = (_Float16*)(ws + 256 + hgBytes);

    zero_accum<<<1, 1, 0, stream>>>(sumS);
    prep_kernel<<<NN, 64, 0, stream>>>(h_tilde, T, alpha, Wp, bp, hK, gK);
    // 65536 tiles, 8 waves/block -> 8192 blocks
    fused_gemm<<<8192, 256, 0, stream>>>(hK, gK, sumS, out, 0);
    fused_gemm<<<8192, 256, 0, stream>>>(hK, gK, sumS, out, 1);
}
